// Basenet_social_15839839387968
// MI455X (gfx1250) — matrix operand order repaired and checked by reading, not compiled, not run on hardware
//
#include <hip/hip_runtime.h>
#include <hip/hip_bf16.h>
#include <cstdint>

typedef __bf16 bf16_t;
typedef __attribute__((ext_vector_type(16))) __bf16 v16bf;
typedef __attribute__((ext_vector_type(8)))  __bf16 v8bf;
typedef __attribute__((ext_vector_type(8)))  float  v8f;
typedef __attribute__((ext_vector_type(4)))  int    v4i;

#define BT_DIM 80
#define NBOX   13
#define SDIM   512
#define EDIM   7
#define MROWS  (BT_DIM * NBOX)   /* 1040 */

#define AS1 __attribute__((address_space(1)))
#define AS3 __attribute__((address_space(3)))

// CDNA5 async global->LDS copy path (ASYNCcnt-tracked). Guarded so the file
// still compiles (synchronous fallback) if this toolchain lacks the builtin.
#if defined(__has_builtin)
#if __has_builtin(__builtin_amdgcn_global_load_async_to_lds_b128) && \
    __has_builtin(__builtin_amdgcn_s_wait_asynccnt)
#define HAVE_ASYNC_LDS 1
#endif
#endif
#ifndef HAVE_ASYNC_LDS
#define HAVE_ASYNC_LDS 0
#endif

#if HAVE_ASYNC_LDS
// Generic->AS1: flat address of a global pointer is the global address.
// (Builtin takes a non-const AS1 pointer; the integer round-trip drops const.)
__device__ __forceinline__ AS1 v4i* as_global_v4i(const void* p) {
    return (AS1 v4i*)(unsigned long long)(uintptr_t)p;
}
// Generic->AS3: low 32 bits of a flat LDS pointer are the LDS offset
// (ISA 10.2: LDS aperture addresses truncate to addr[31:0]).
__device__ __forceinline__ AS3 v4i* as_lds_v4i(void* p) {
    return (AS3 v4i*)(unsigned int)(uintptr_t)p;
}
#endif

// ---------------------------------------------------------------------------
// Pre-pass: convert f32 weights [K][N] -> bf16 transposed [N][K].
// Tiled through LDS so both global reads and writes stay coalesced.
// K, N multiples of 32 for all call sites. Batched over blockIdx.z.
// ---------------------------------------------------------------------------
__launch_bounds__(256)
__global__ void transpose_cvt_kernel(const float* __restrict__ in,
                                     bf16_t* __restrict__ out,
                                     int K, int N, long sIn, long sOut)
{
    __shared__ float tile[32][33];
    in  += (long)blockIdx.z * sIn;
    out += (long)blockIdx.z * sOut;
    const int k0 = blockIdx.y * 32, n0 = blockIdx.x * 32;
    const int tx = threadIdx.x & 31, ty = threadIdx.x >> 5;   // 32 x 8
    #pragma unroll
    for (int r = ty; r < 32; r += 8)
        tile[r][tx] = in[(long)(k0 + r) * N + (n0 + tx)];
    __syncthreads();
    #pragma unroll
    for (int r = ty; r < 32; r += 8)
        out[(long)(n0 + r) * K + (k0 + tx)] = (bf16_t)tile[tx][r];
}

// ---------------------------------------------------------------------------
// GEMM through bf16 WMMA, f32 accumulate:  C = act(A @ B + bias)
// A:  Mh x Kh f32 (row stride lda), staged + converted to bf16 via LDS.
// Bt: pre-converted bf16 weights in [N][K] layout (row stride Kh) -> the
//     B-stage is a pure 16B/lane copy, done with GLOBAL_LOAD_ASYNC_TO_LDS_B128
//     when available.
// Ping-pong LDS double buffering: one barrier per 32-wide K step.
// WG tile 128(M) x 64(N), 8 waves, wave tile 32x32 = 2x2 wmma accumulators.
// act: 0=none, 1=sigmoid, 2=tanh.  Kh % 32 == 0, Nh % 64 == 0.
// ---------------------------------------------------------------------------
__launch_bounds__(256)
__global__ void gemm_bf16_wmma(const float* __restrict__ A,
                               const bf16_t* __restrict__ Bt,
                               const float* __restrict__ bias,
                               float* __restrict__ C,
                               int Mh, int Nh, int Kh,
                               int lda, int ldc,
                               long sA, long sB, long sBias, long sC,
                               int act)
{
    // stride 40 bf16 (80B rows): keeps all b128 LDS ops 16B-aligned and
    // fragment reads bank-conflict-free (20*row mod 64 distinct over 16 rows).
    __shared__ bf16_t Alds[2][128][40];
    __shared__ bf16_t Blds[2][64][40];

    const int z = blockIdx.z;
    A  += (long)z * sA;
    Bt += (long)z * sB;
    C  += (long)z * sC;
    const float* biasp = bias ? (bias + (long)z * sBias) : nullptr;

    const int tid  = threadIdx.x;
    const int lane = tid & 31;
    const int wave = tid >> 5;
    const int wm   = wave & 3;          // 0..3 along M
    const int wn   = wave >> 2;         // 0..1 along N
    const int mBase = blockIdx.y * 128;
    const int nBase = blockIdx.x * 64;

    // A staging: thread -> (row 0..127, 16-float strip)
    const int ar  = tid >> 1;
    const int ac0 = (tid & 1) * 16;
    // B staging: thread -> (col 0..63, 8-bf16 k-strip) : exactly 16B per lane
    const int bn  = tid >> 2;
    const int bk0 = (tid & 3) * 8;

    const int mrow = lane & 15;
    const int hi   = lane >> 4;

    // rows >= Mh are never stored by the epilogue -> clamp instead of branch
    const int grow = (mBase + ar < Mh) ? (mBase + ar) : (Mh - 1);
    const float*  aRow = A + (long)grow * lda + ac0;
    const bf16_t* bRow = Bt + (long)(nBase + bn) * Kh + bk0;

    v8f acc[2][2] = {};
    const int nT = Kh >> 5;

    float4 areg[4];
#if !HAVE_ASYNC_LDS
    uint4 breg;
#endif

    // ---- prologue: stage tile 0 into buffer 0 ----
    {
        #pragma unroll
        for (int j = 0; j < 4; ++j) areg[j] = *(const float4*)(aRow + 4 * j);
#if HAVE_ASYNC_LDS
        __builtin_amdgcn_global_load_async_to_lds_b128(
            as_global_v4i(bRow), as_lds_v4i(&Blds[0][bn][bk0]), 0, 0);
#else
        breg = *(const uint4*)bRow;
        *(uint4*)&Blds[0][bn][bk0] = breg;
#endif
        v8bf p0, p1;
        #pragma unroll
        for (int j = 0; j < 4; ++j) {
            p0[2*j]   = (bf16_t)areg[j].x;  p0[2*j+1] = (bf16_t)areg[j].y;
            p1[2*j]   = (bf16_t)areg[j].z;  p1[2*j+1] = (bf16_t)areg[j].w;
        }
        // interleave back into memory order x,y,z,w per float4
        v8bf lo  = __builtin_shufflevector(p0, p1, 0, 1, 8, 9, 2, 3, 10, 11);
        v8bf hic = __builtin_shufflevector(p0, p1, 4, 5, 12, 13, 6, 7, 14, 15);
        *(v8bf*)&Alds[0][ar][ac0]     = lo;
        *(v8bf*)&Alds[0][ar][ac0 + 8] = hic;
#if HAVE_ASYNC_LDS
        __builtin_amdgcn_s_wait_asynccnt(0);
#endif
    }
    __syncthreads();

    for (int t = 0; t < nT; ++t) {
        const int cur = t & 1, nxt = cur ^ 1;
        const bool more = (t + 1) < nT;

        // ---- issue next tile's global traffic before computing ----
        if (more) {
            const float* ap = aRow + ((t + 1) << 5);
            #pragma unroll
            for (int j = 0; j < 4; ++j) areg[j] = *(const float4*)(ap + 4 * j);
#if HAVE_ASYNC_LDS
            __builtin_amdgcn_global_load_async_to_lds_b128(
                as_global_v4i(bRow + ((t + 1) << 5)),
                as_lds_v4i(&Blds[nxt][bn][bk0]), 0, 0);
#else
            breg = *(const uint4*)(bRow + ((t + 1) << 5));
#endif
        }

        // ---- compute tile t (documented wave32 16-bit fragment layouts) ----
        v16bf afrag[2], bfrag[2];
        #pragma unroll
        for (int mi = 0; mi < 2; ++mi) {
            const v8bf* ap = (const v8bf*)&Alds[cur][wm * 32 + mi * 16 + mrow][hi * 8];
            v8bf l = ap[0], h = ap[2];   // K = kb+0..7 and kb+16..23
            afrag[mi] = __builtin_shufflevector(l, h, 0,1,2,3,4,5,6,7,8,9,10,11,12,13,14,15);
        }
        #pragma unroll
        for (int ni = 0; ni < 2; ++ni) {
            const v8bf* bp = (const v8bf*)&Blds[cur][wn * 32 + ni * 16 + mrow][hi * 8];
            v8bf l = bp[0], h = bp[2];   // K = kb+0..7 and kb+16..23
            bfrag[ni] = __builtin_shufflevector(l, h, 0,1,2,3,4,5,6,7,8,9,10,11,12,13,14,15);
        }
        #pragma unroll
        for (int mi = 0; mi < 2; ++mi)
            #pragma unroll
            for (int ni = 0; ni < 2; ++ni)
                acc[mi][ni] = __builtin_amdgcn_wmma_f32_16x16x32_bf16(
                    false, afrag[mi], false, bfrag[ni],
                    (short)0, acc[mi][ni], false, false);

        // ---- finish staging tile t+1 into the other buffer ----
        if (more) {
            v8bf p0, p1;
            #pragma unroll
            for (int j = 0; j < 4; ++j) {
                p0[2*j]   = (bf16_t)areg[j].x;  p0[2*j+1] = (bf16_t)areg[j].y;
                p1[2*j]   = (bf16_t)areg[j].z;  p1[2*j+1] = (bf16_t)areg[j].w;
            }
            v8bf lo  = __builtin_shufflevector(p0, p1, 0, 1, 8, 9, 2, 3, 10, 11);
            v8bf hic = __builtin_shufflevector(p0, p1, 4, 5, 12, 13, 6, 7, 14, 15);
            *(v8bf*)&Alds[nxt][ar][ac0]     = lo;
            *(v8bf*)&Alds[nxt][ar][ac0 + 8] = hic;
#if !HAVE_ASYNC_LDS
            *(uint4*)&Blds[nxt][bn][bk0] = breg;
#endif
#if HAVE_ASYNC_LDS
            __builtin_amdgcn_s_wait_asynccnt(0);
#endif
            __syncthreads();
        }
    }

    // ---- epilogue: C/D layout VGPR v -> M = v + 8*laneHalf, N = lane&15 ----
    #pragma unroll
    for (int ni = 0; ni < 2; ++ni) {
        const int gn = nBase + wn * 32 + ni * 16 + mrow;
        const float bv = biasp ? biasp[gn] : 0.f;
        #pragma unroll
        for (int mi = 0; mi < 2; ++mi) {
            #pragma unroll
            for (int v = 0; v < 8; ++v) {
                const int gm = mBase + wm * 32 + mi * 16 + v + hi * 8;
                if (gm < Mh) {
                    float x = acc[mi][ni][v] + bv;
                    if (act == 1)      x = 1.f / (1.f + __expf(-x));
                    else if (act == 2) x = tanhf(x);
                    C[(long)gm * ldc + gn] = x;
                }
            }
        }
    }
}

// ---------------------------------------------------------------------------
// row[r,e] = msgs[r,e,:] . attn_Wrow[e,:]; col likewise. One wave per (r,e).
// ---------------------------------------------------------------------------
__launch_bounds__(256)
__global__ void rowcol_kernel(const float* __restrict__ msgs,
                              const float* __restrict__ Wrow,
                              const float* __restrict__ Wcol,
                              float* __restrict__ rowv, float* __restrict__ colv)
{
    const int idx = blockIdx.x * 8 + (threadIdx.x >> 5);   // (r*7 + e)
    if (idx >= MROWS * EDIM) return;
    const int e    = idx % EDIM;
    const int lane = threadIdx.x & 31;
    const float* m  = msgs + (long)idx * SDIM;
    const float* wr = Wrow + (long)e * SDIM;
    const float* wc = Wcol + (long)e * SDIM;
    float sr = 0.f, sc = 0.f;
    for (int d = lane; d < SDIM; d += 32) {
        const float v = m[d];
        sr += v * wr[d];
        sc += v * wc[d];
    }
    #pragma unroll
    for (int off = 16; off > 0; off >>= 1) {
        sr += __shfl_down(sr, off, 32);
        sc += __shfl_down(sc, off, 32);
    }
    if (lane == 0) { rowv[idx] = sr; colv[idx] = sc; }
}

// ---------------------------------------------------------------------------
// Fused: scores = sigmoid(row+col+b); merged = scores @ msgs;
// abuf = concat(merged, prop). One block per destination row r1 = bt*13+m.
// ---------------------------------------------------------------------------
__launch_bounds__(256)
__global__ void scores_merge_kernel(const float* __restrict__ msgs,
                                    const float* __restrict__ rowv,
                                    const float* __restrict__ colv,
                                    const float* __restrict__ attnB,
                                    const float* __restrict__ prop,
                                    float* __restrict__ scores,
                                    float* __restrict__ abuf)
{
    __shared__ float s[NBOX * EDIM];  // 91
    const int r1 = blockIdx.x;        // bt*13 + m
    const int bt = r1 / NBOX;
    const int t  = threadIdx.x;
    if (t < NBOX * EDIM) {
        const int n2 = t / EDIM, e = t % EDIM;
        const float x = rowv[r1 * EDIM + e] + colv[(bt * NBOX + n2) * EDIM + e] + attnB[e];
        const float v = 1.f / (1.f + __expf(-x));
        s[t] = v;
        scores[(long)r1 * (NBOX * EDIM) + t] = v;   // [(bt,m),n2,e]
    }
    __syncthreads();
    const float* base = msgs + (long)bt * NBOX * EDIM * SDIM;
    for (int d = t; d < SDIM; d += 256) {
        float acc = 0.f;
        #pragma unroll 7
        for (int j = 0; j < NBOX * EDIM; ++j)
            acc += s[j] * base[(long)j * SDIM + d];
        abuf[(long)r1 * (2 * SDIM) + d]        = acc;                       // merged
        abuf[(long)r1 * (2 * SDIM) + SDIM + d] = prop[(long)r1 * SDIM + d]; // prop
    }
}

// a2 = concat(merged, r * prop)
__launch_bounds__(256)
__global__ void a2_kernel(const float* __restrict__ abuf,
                          const float* __restrict__ rbuf,
                          const float* __restrict__ prop,
                          float* __restrict__ a2)
{
    const int i = blockIdx.x * 256 + threadIdx.x;
    if (i >= MROWS * SDIM) return;
    const int r = i >> 9, d = i & (SDIM - 1);
    a2[(long)r * (2 * SDIM) + d]        = abuf[(long)r * (2 * SDIM) + d];
    a2[(long)r * (2 * SDIM) + SDIM + d] = rbuf[i] * prop[i];
}

// prop = (1-z)*prop + z*h_hat
__launch_bounds__(256)
__global__ void prop_update_kernel(float* __restrict__ prop,
                                   const float* __restrict__ zbuf,
                                   const float* __restrict__ hbuf)
{
    const int i = blockIdx.x * 256 + threadIdx.x;
    if (i >= MROWS * SDIM) return;
    const float z = zbuf[i];
    prop[i] = (1.f - z) * prop[i] + z * hbuf[i];
}

// ---------------------------------------------------------------------------
// Final reductions: action_score, actor_score, acty softmax. One block per bt.
// ---------------------------------------------------------------------------
__launch_bounds__(256)
__global__ void outputs_kernel(const float* __restrict__ scores,
                               float* __restrict__ out)
{
    __shared__ float actor_s[NBOX * NBOX];
    __shared__ float action_s[NBOX * EDIM];
    __shared__ float asum[NBOX];
    const int bt = blockIdx.x;
    const float* sc = scores + (long)bt * NBOX * NBOX * EDIM;
    const int t = threadIdx.x;

    if (t < NBOX * NBOX) {
        const float a = 1.f - sc[t * EDIM + 0];
        actor_s[t] = a;
        out[7280 + 480 + (long)bt * (NBOX * NBOX) + t] = a;    // actor_score
    }
    __syncthreads();
    if (t < NBOX * EDIM) {
        const int n1 = t / EDIM, e = t % EDIM;
        float acc = 0.f;
        #pragma unroll
        for (int n2 = 0; n2 < NBOX; ++n2)
            acc += actor_s[n1 * NBOX + n2] * sc[(n1 * NBOX + n2) * EDIM + e];
        action_s[t] = acc;
        out[(long)bt * (NBOX * EDIM) + t] = acc;               // action_score
    }
    if (t < NBOX) {
        float ssum = 0.f;
        #pragma unroll
        for (int n2 = 0; n2 < NBOX; ++n2) ssum += actor_s[t * NBOX + n2];
        asum[t] = ssum;
    }
    __syncthreads();
    if (t == 0) {
        float tv[EDIM];
        #pragma unroll
        for (int e = 0; e < EDIM; ++e) {
            float svv = 0.f;
            for (int n1 = 0; n1 < NBOX; ++n1)
                svv += action_s[n1 * EDIM + e] * asum[n1];
            tv[e] = svv;
        }
        float mx = tv[1];
        for (int e = 2; e < EDIM; ++e) mx = fmaxf(mx, tv[e]);
        float ex[EDIM - 1], den = 0.f;
        for (int e = 1; e < EDIM; ++e) { ex[e - 1] = __expf(tv[e] - mx); den += ex[e - 1]; }
        for (int e = 1; e < EDIM; ++e)
            out[7280 + bt * (EDIM - 1) + (e - 1)] = ex[e - 1] / den;       // acty_score
    }
}

// ---------------------------------------------------------------------------
extern "C" void kernel_launch(void* const* d_in, const int* in_sizes, int n_in,
                              void* d_out, int out_size, void* d_ws, size_t ws_size,
                              hipStream_t stream)
{
    const float* boxes  = (const float*)d_in[0];
    const float* compW  = (const float*)d_in[1];
    const float* compB  = (const float*)d_in[2];
    const float* edgeW  = (const float*)d_in[3];
    const float* edgeB  = (const float*)d_in[4];
    const float* attnWr = (const float*)d_in[5];
    const float* attnWc = (const float*)d_in[6];
    const float* attnB  = (const float*)d_in[7];
    const float* resetW = (const float*)d_in[8];
    const float* resetB = (const float*)d_in[9];
    const float* updW   = (const float*)d_in[10];
    const float* updB   = (const float*)d_in[11];
    const float* trW    = (const float*)d_in[12];
    const float* trB    = (const float*)d_in[13];
    float* out = (float*)d_out;

    char* p = (char*)d_ws;
    auto alloc = [&](size_t bytes) -> void* {
        void* r = (void*)p;
        p += (bytes + 255) & ~(size_t)255;
        return r;
    };
    float*  prop   = (float*)alloc((size_t)MROWS * SDIM * 4);
    float*  msgs   = (float*)alloc((size_t)MROWS * EDIM * SDIM * 4);
    float*  rowv   = (float*)alloc((size_t)MROWS * EDIM * 4);
    float*  colv   = (float*)alloc((size_t)MROWS * EDIM * 4);
    float*  scores = (float*)alloc((size_t)MROWS * NBOX * EDIM * 4);
    float*  abuf   = (float*)alloc((size_t)MROWS * 2 * SDIM * 4);
    float*  a2buf  = (float*)alloc((size_t)MROWS * 2 * SDIM * 4);
    float*  rbuf   = (float*)alloc((size_t)MROWS * SDIM * 4);
    float*  zbuf   = (float*)alloc((size_t)MROWS * SDIM * 4);
    float*  hbuf   = (float*)alloc((size_t)MROWS * SDIM * 4);
    bf16_t* wtC    = (bf16_t*)alloc((size_t)26400 * SDIM * 2);
    bf16_t* wtE    = (bf16_t*)alloc((size_t)EDIM * SDIM * SDIM * 2);
    bf16_t* wtR    = (bf16_t*)alloc((size_t)2 * SDIM * SDIM * 2);
    bf16_t* wtU    = (bf16_t*)alloc((size_t)2 * SDIM * SDIM * 2);
    bf16_t* wtT    = (bf16_t*)alloc((size_t)2 * SDIM * SDIM * 2);

    const dim3 blk(256);
    const int gM = (MROWS + 127) / 128;   // 9
    const int gN = SDIM / 64;             // 8

    // ---- 0) one-shot weight convert+transpose to bf16 [N][K] ----
    transpose_cvt_kernel<<<dim3(SDIM / 32, 26400 / 32, 1), blk, 0, stream>>>(
        compW, wtC, 26400, SDIM, 0, 0);
    transpose_cvt_kernel<<<dim3(SDIM / 32, SDIM / 32, EDIM), blk, 0, stream>>>(
        edgeW, wtE, SDIM, SDIM, (long)SDIM * SDIM, (long)SDIM * SDIM);
    transpose_cvt_kernel<<<dim3(SDIM / 32, (2 * SDIM) / 32, 1), blk, 0, stream>>>(
        resetW, wtR, 2 * SDIM, SDIM, 0, 0);
    transpose_cvt_kernel<<<dim3(SDIM / 32, (2 * SDIM) / 32, 1), blk, 0, stream>>>(
        updW, wtU, 2 * SDIM, SDIM, 0, 0);
    transpose_cvt_kernel<<<dim3(SDIM / 32, (2 * SDIM) / 32, 1), blk, 0, stream>>>(
        trW, wtT, 2 * SDIM, SDIM, 0, 0);

    // ---- 1) compress: prop = boxes @ compW + compB ----
    gemm_bf16_wmma<<<dim3(gN, gM, 1), blk, 0, stream>>>(
        boxes, wtC, compB, prop, MROWS, SDIM, 26400,
        26400, SDIM, 0, 0, 0, 0, /*act*/0);

    for (int it = 0; it < 2; ++it) {
        // 2) msgs[e] = prop @ edgeW[e] + edgeB[e]  (batched over e)
        gemm_bf16_wmma<<<dim3(gN, gM, EDIM), blk, 0, stream>>>(
            prop, wtE, edgeB, msgs, MROWS, SDIM, SDIM,
            SDIM, EDIM * SDIM,
            /*sA*/0, /*sB*/(long)SDIM * SDIM, /*sBias*/SDIM, /*sC*/SDIM, /*act*/0);
        // 3) attention row/col dot products
        rowcol_kernel<<<dim3((MROWS * EDIM + 7) / 8), blk, 0, stream>>>(
            msgs, attnWr, attnWc, rowv, colv);
        // 4) scores + merged + concat(merged, prop)
        scores_merge_kernel<<<dim3(MROWS), blk, 0, stream>>>(
            msgs, rowv, colv, attnB, prop, scores, abuf);
        // 5) r = sigmoid(a @ resetW + b); z = sigmoid(a @ updW + b)
        gemm_bf16_wmma<<<dim3(gN, gM, 1), blk, 0, stream>>>(
            abuf, wtR, resetB, rbuf, MROWS, SDIM, 2 * SDIM,
            2 * SDIM, SDIM, 0, 0, 0, 0, /*act*/1);
        gemm_bf16_wmma<<<dim3(gN, gM, 1), blk, 0, stream>>>(
            abuf, wtU, updB, zbuf, MROWS, SDIM, 2 * SDIM,
            2 * SDIM, SDIM, 0, 0, 0, 0, /*act*/1);
        // 6) a2 = concat(merged, r*prop)
        a2_kernel<<<dim3((MROWS * SDIM + 255) / 256), blk, 0, stream>>>(
            abuf, rbuf, prop, a2buf);
        // 7) h_hat = tanh(a2 @ trW + b)
        gemm_bf16_wmma<<<dim3(gN, gM, 1), blk, 0, stream>>>(
            a2buf, wtT, trB, hbuf, MROWS, SDIM, 2 * SDIM,
            2 * SDIM, SDIM, 0, 0, 0, 0, /*act*/2);
        // 8) prop = (1-z)*prop + z*h_hat
        prop_update_kernel<<<dim3((MROWS * SDIM + 255) / 256), blk, 0, stream>>>(
            prop, zbuf, hbuf);
    }

    // 9) outputs from final scores
    outputs_kernel<<<dim3(BT_DIM), blk, 0, stream>>>(scores, out);

    (void)in_sizes; (void)n_in; (void)out_size; (void)ws_size;
}
